// MultiHeadAttention_14903536517723
// MI455X (gfx1250) — matrix operand order repaired and checked by reading, not compiled, not run on hardware
//
#include <hip/hip_runtime.h>
#include <hip/hip_bf16.h>

// ---------------------------------------------------------------------------
// MHA for MI455X (gfx1250): bf16 WMMA GEMMs + LDS softmax attention.
// Workspace usage: 48 MB (Xbf 8 + W*4 bf16 8 + Q 8 + K 8 + V^T 8 + attn 8).
// ---------------------------------------------------------------------------

typedef __bf16 bf16_t;
typedef __attribute__((ext_vector_type(16))) __bf16 v16bf;
typedef __attribute__((ext_vector_type(8)))  __bf16 v8bf;
typedef __attribute__((ext_vector_type(4)))  __bf16 v4bf;
typedef __attribute__((ext_vector_type(8)))  float  v8f;
typedef __attribute__((ext_vector_type(4)))  float  v4f;

#define EMBED 1024
#define SEQ   2048
#define BATCH 2
#define HEADS 16
#define HDIM  64
#define MROWS (BATCH*SEQ)      // 4096
#define SSTRIDE 2056           // padded LDS row stride (floats)

// A-operand fragment (16x32, 16-bit): lane holds row (l&15); vector elems
// 0..7  = K[hi*8   .. hi*8+7], elems 8..15 = K[16+hi*8 .. 16+hi*8+7].
static __device__ inline v16bf load_afrag(const bf16_t* rowptr, int kbase, int hi) {
  union { v16bf v; v8bf h[2]; } u;
  const bf16_t* p = rowptr + kbase + hi * 8;
  u.h[0] = *(const v8bf*)(p);
  u.h[1] = *(const v8bf*)(p + 16);
  return u.v;
}

// B-operand fragment (32x16, 16-bit): lane holds column (l&15); vector elems
// 0..7  = K[hi*8   .. hi*8+7], elems 8..15 = K[16+hi*8 .. 16+hi*8+7].
static __device__ inline v16bf load_bfrag(const bf16_t* colptr, int kbase, int hi) {
  union { v16bf v; v8bf h[2]; } u;
  const bf16_t* p = colptr + kbase + hi * 8;
  u.h[0] = *(const v8bf*)(p);
  u.h[1] = *(const v8bf*)(p + 16);
  return u.v;
}

static __device__ inline v8f wmma_bf16(v16bf a, v16bf b, v8f c) {
  return __builtin_amdgcn_wmma_f32_16x16x32_bf16(
      /*neg_a=*/false, a, /*neg_b=*/false, b,
      /*c_mod=*/(short)0, c, /*reuse_a=*/false, /*reuse_b=*/false);
}

// ---------------------------------------------------------------------------
// f32 -> bf16 conversion, 4 elements per thread
// ---------------------------------------------------------------------------
__global__ __launch_bounds__(256)
void cvt_kernel(const float* __restrict__ in, bf16_t* __restrict__ out) {
  size_t i = ((size_t)blockIdx.x * blockDim.x + threadIdx.x) * 4;
  v4f f = *(const v4f*)(in + i);
  v4bf o;
  o[0] = (bf16_t)f[0]; o[1] = (bf16_t)f[1]; o[2] = (bf16_t)f[2]; o[3] = (bf16_t)f[3];
  *(v4bf*)(out + i) = o;
}

// ---------------------------------------------------------------------------
// GEMM: C[M=4096, N=1024] = A[M,1024] * W[N,1024]^T + bias
// Each wave: 32(M) x 64(N) tile. MODE: 0=Q(out bf16 [B,H,S,D], *0.125)
// 1=K(bf16 [B,H,S,D])  2=V(bf16 [B,H,D,S] transposed)  3=O(f32 row-major)
// ---------------------------------------------------------------------------
template <int MODE>
__global__ __launch_bounds__(256)
void gemm_kernel(const bf16_t* __restrict__ A, const bf16_t* __restrict__ W,
                 const float* __restrict__ bias, void* __restrict__ out) {
  const int lane = threadIdx.x & 31;
  const int hi = lane >> 4;
  const int lo = lane & 15;
  const int widx = blockIdx.x * 8 + (threadIdx.x >> 5);
  const int m0 = (widx >> 4) * 32;   // 128 m-tiles of 32
  const int n0 = (widx & 15) * 64;   // 16 n-tiles of 64

  const v8f vzero = {0.f, 0.f, 0.f, 0.f, 0.f, 0.f, 0.f, 0.f};
  v8f acc[2][4];
#pragma unroll
  for (int t = 0; t < 2; ++t)
#pragma unroll
    for (int j = 0; j < 4; ++j) acc[t][j] = vzero;

  const bf16_t* arow0 = A + (size_t)(m0 + lo) * EMBED;
  const bf16_t* arow1 = A + (size_t)(m0 + 16 + lo) * EMBED;
  const bf16_t* wrow[4];
#pragma unroll
  for (int j = 0; j < 4; ++j) wrow[j] = W + (size_t)(n0 + j * 16 + lo) * EMBED;

  for (int k = 0; k < EMBED; k += 32) {
    v16bf a0 = load_afrag(arow0, k, hi);
    v16bf a1 = load_afrag(arow1, k, hi);
#pragma unroll
    for (int j = 0; j < 4; ++j) {
      v16bf b = load_bfrag(wrow[j], k, hi);
      acc[0][j] = wmma_bf16(a0, b, acc[0][j]);
      acc[1][j] = wmma_bf16(a1, b, acc[1][j]);
    }
  }

#pragma unroll
  for (int j = 0; j < 4; ++j) {
    const int n = n0 + j * 16 + lo;
    const float bv = bias[n];
#pragma unroll
    for (int t = 0; t < 2; ++t) {
#pragma unroll
      for (int r = 0; r < 8; ++r) {
        const int m = m0 + t * 16 + r + 8 * hi;
        const float val = acc[t][j][r] + bv;
        if (MODE == 3) {
          ((float*)out)[(size_t)m * EMBED + n] = val;
        } else {
          const int b = m >> 11, s = m & (SEQ - 1);
          const int h = n >> 6, d = n & (HDIM - 1);
          if (MODE == 0) {
            ((bf16_t*)out)[(((size_t)b * HEADS + h) * SEQ + s) * HDIM + d] =
                (bf16_t)(val * 0.125f);  // 1/sqrt(64) folded into Q
          } else if (MODE == 1) {
            ((bf16_t*)out)[(((size_t)b * HEADS + h) * SEQ + s) * HDIM + d] = (bf16_t)val;
          } else {
            ((bf16_t*)out)[(((size_t)b * HEADS + h) * HDIM + d) * SEQ + s] = (bf16_t)val;
          }
        }
      }
    }
  }
}

// ---------------------------------------------------------------------------
// Attention: one block (4 waves, 128 thr) per (b, h, 16-query block).
// LDS: full 16 x 2048 score row block, two-pass softmax, then P*V via WMMA.
// ---------------------------------------------------------------------------
__global__ __launch_bounds__(128)
void attn_kernel(const bf16_t* __restrict__ Qb, const bf16_t* __restrict__ Kb,
                 const bf16_t* __restrict__ VTb, bf16_t* __restrict__ attn) {
  extern __shared__ float smem[];
  float* scores = smem;                   // 16 * SSTRIDE
  float* red    = smem + 16 * SSTRIDE;    // 128
  float* red2   = red + 128;              // 128
  float* rowinv = red2 + 128;             // 16

  const int tid = threadIdx.x;
  const int lane = tid & 31;
  const int w = tid >> 5;                 // wave 0..3
  const int hi = lane >> 4, lo = lane & 15;

  const int bid = blockIdx.x;
  const int qblk = bid & 127;             // SEQ/16
  const int h = (bid >> 7) & (HEADS - 1);
  const int b = bid >> 11;
  const int q0 = qblk * 16;

  const bf16_t* Qh = Qb + ((size_t)b * HEADS + h) * SEQ * HDIM;
  const bf16_t* Kh = Kb + ((size_t)b * HEADS + h) * SEQ * HDIM;
  const bf16_t* VT = VTb + ((size_t)b * HEADS + h) * HDIM * SEQ;

  const v8f vzero = {0.f, 0.f, 0.f, 0.f, 0.f, 0.f, 0.f, 0.f};

  // Q fragments (rows q0..q0+15, D=64 -> two K-steps); each wave loads a copy.
  const bf16_t* qrow = Qh + (size_t)(q0 + lo) * HDIM;
  const v16bf aQ0 = load_afrag(qrow, 0, hi);
  const v16bf aQ1 = load_afrag(qrow, 32, hi);

  // ---- Phase 1: scores (Q/8) . K^T for key strip [w*512, w*512+512) ----
  for (int kt = 0; kt < 32; ++kt) {
    const int key0 = w * 512 + kt * 16;
    const bf16_t* krow = Kh + (size_t)(key0 + lo) * HDIM;
    v16bf bK0 = load_bfrag(krow, 0, hi);
    v16bf bK1 = load_bfrag(krow, 32, hi);
    v8f c = vzero;
    c = wmma_bf16(aQ0, bK0, c);
    c = wmma_bf16(aQ1, bK1, c);
#pragma unroll
    for (int r = 0; r < 8; ++r)
      scores[(r + 8 * hi) * SSTRIDE + key0 + lo] = c[r];
  }
  __syncthreads();

  // ---- Phase 2: softmax over 2048 keys per row (8 threads per row) ----
  {
    const int row = tid >> 3, ch = tid & 7;
    float* srow = scores + row * SSTRIDE;
    const int c0 = ch * 256;
    float mx = -3.0e38f;
    for (int c = 0; c < 256; ++c) mx = fmaxf(mx, srow[c0 + c]);
    red[row * 8 + ch] = mx;
    __syncthreads();
    float m = red[row * 8];
#pragma unroll
    for (int i = 1; i < 8; ++i) m = fmaxf(m, red[row * 8 + i]);
    float sum = 0.f;
    for (int c = 0; c < 256; ++c) {
      float e = __expf(srow[c0 + c] - m);
      srow[c0 + c] = e;
      sum += e;
    }
    red2[row * 8 + ch] = sum;
    __syncthreads();
    if (ch == 0) {
      float s = 0.f;
#pragma unroll
      for (int i = 0; i < 8; ++i) s += red2[row * 8 + i];
      rowinv[row] = 1.0f / s;
    }
  }
  __syncthreads();

  // ---- Phase 3: O(16q x 16d) = P(16x2048) . V(2048 x 16d), d in [w*16, +16) ----
  v8f acc = vzero;
  const bf16_t* vrow = VT + (size_t)(w * 16 + lo) * SEQ;
  const float* prow = scores + (size_t)lo * SSTRIDE;
  for (int kt = 0; kt < 64; ++kt) {
    const float* sp = prow + kt * 32 + hi * 8;
    v4f f0 = *(const v4f*)(sp);
    v4f f1 = *(const v4f*)(sp + 4);
    v4f f2 = *(const v4f*)(sp + 16);
    v4f f3 = *(const v4f*)(sp + 20);
    v16bf a;
    a[0]  = (bf16_t)f0[0]; a[1]  = (bf16_t)f0[1]; a[2]  = (bf16_t)f0[2]; a[3]  = (bf16_t)f0[3];
    a[4]  = (bf16_t)f1[0]; a[5]  = (bf16_t)f1[1]; a[6]  = (bf16_t)f1[2]; a[7]  = (bf16_t)f1[3];
    a[8]  = (bf16_t)f2[0]; a[9]  = (bf16_t)f2[1]; a[10] = (bf16_t)f2[2]; a[11] = (bf16_t)f2[3];
    a[12] = (bf16_t)f3[0]; a[13] = (bf16_t)f3[1]; a[14] = (bf16_t)f3[2]; a[15] = (bf16_t)f3[3];
    v16bf bV = load_bfrag(vrow, kt * 32, hi);
    acc = wmma_bf16(a, bV, acc);
  }

#pragma unroll
  for (int r = 0; r < 8; ++r) {
    const int q = r + 8 * hi;
    const float val = acc[r] * rowinv[q];
    const int s = q0 + q;
    const int d = w * 16 + lo;
    attn[((size_t)b * SEQ + s) * EMBED + h * HDIM + d] = (bf16_t)val;
  }
}

// ---------------------------------------------------------------------------
extern "C" void kernel_launch(void* const* d_in, const int* in_sizes, int n_in,
                              void* d_out, int out_size, void* d_ws, size_t ws_size,
                              hipStream_t stream) {
  (void)in_sizes; (void)n_in; (void)out_size; (void)ws_size;

  const float* X  = (const float*)d_in[0];
  const float* Wq = (const float*)d_in[1];
  const float* bq = (const float*)d_in[2];
  const float* Wk = (const float*)d_in[3];
  const float* bk = (const float*)d_in[4];
  const float* Wv = (const float*)d_in[5];
  const float* bv = (const float*)d_in[6];
  const float* Wo = (const float*)d_in[7];
  const float* bo = (const float*)d_in[8];

  char* ws = (char*)d_ws;
  bf16_t* Xbf  = (bf16_t*)ws; ws += (size_t)MROWS * EMBED * 2;
  bf16_t* Wqb  = (bf16_t*)ws; ws += (size_t)EMBED * EMBED * 2;
  bf16_t* Wkb  = (bf16_t*)ws; ws += (size_t)EMBED * EMBED * 2;
  bf16_t* Wvb  = (bf16_t*)ws; ws += (size_t)EMBED * EMBED * 2;
  bf16_t* Wob  = (bf16_t*)ws; ws += (size_t)EMBED * EMBED * 2;
  bf16_t* Qbuf = (bf16_t*)ws; ws += (size_t)MROWS * EMBED * 2;   // [B,H,S,D]
  bf16_t* Kbuf = (bf16_t*)ws; ws += (size_t)MROWS * EMBED * 2;   // [B,H,S,D]
  bf16_t* VTb  = (bf16_t*)ws; ws += (size_t)MROWS * EMBED * 2;   // [B,H,D,S]
  bf16_t* Attn = (bf16_t*)ws; ws += (size_t)MROWS * EMBED * 2;   // [B,S,E]

  // f32 -> bf16 conversions
  cvt_kernel<<<(MROWS * EMBED) / 1024, 256, 0, stream>>>(X, Xbf);
  cvt_kernel<<<(EMBED * EMBED) / 1024, 256, 0, stream>>>(Wq, Wqb);
  cvt_kernel<<<(EMBED * EMBED) / 1024, 256, 0, stream>>>(Wk, Wkb);
  cvt_kernel<<<(EMBED * EMBED) / 1024, 256, 0, stream>>>(Wv, Wvb);
  cvt_kernel<<<(EMBED * EMBED) / 1024, 256, 0, stream>>>(Wo, Wob);

  // Q/K/V projections (2048 waves = 256 blocks x 8 waves)
  gemm_kernel<0><<<256, 256, 0, stream>>>(Xbf, Wqb, bq, (void*)Qbuf);
  gemm_kernel<1><<<256, 256, 0, stream>>>(Xbf, Wkb, bk, (void*)Kbuf);
  gemm_kernel<2><<<256, 256, 0, stream>>>(Xbf, Wvb, bv, (void*)VTb);

  // Attention: B*H*(S/16) = 4096 blocks, 128 threads, ~130 KB dynamic LDS
  const size_t lds_bytes = (size_t)(16 * SSTRIDE + 128 + 128 + 16) * sizeof(float);
  attn_kernel<<<4096, 128, lds_bytes, stream>>>(Qbuf, Kbuf, VTb, Attn);

  // Output projection -> f32 d_out
  gemm_kernel<3><<<256, 256, 0, stream>>>(Attn, Wob, bo, d_out);
}
